// Seq2SeqCopyPointer_15238543966279
// MI455X (gfx1250) — matrix operand order repaired and checked by reading, not compiled, not run on hardware
//
#include <hip/hip_runtime.h>
#include <hip/hip_bf16.h>
#include <math.h>

// ---------------------------------------------------------------------------
// Seq2Seq copy-pointer transformer forward for gfx1250 (MI455X).
//
// Bandwidth-bound workload (~425MB fp32 weights/pass at 23.3 TB/s). fp32 in
// HBM, on-the-fly fp32->f16 conversion (v_cvt_pkrtz_f16_f32) into ping-pong
// LDS tiles, v_wmma_f32_16x16x32_f16 (f32 accumulate) for every GEMM, fused
// bias/GELU epilogue.
//
// GEMM is templated <GUARD, TRANSB>:
//   - fast path (tile-aligned dims): float4 staging loads -> global_load_b128,
//     uint2 LDS stores -> ds_store_b64, zero exec-mask branching.
//   - guard path (ragged decoder tiles): unconditional loads from clamped
//     addresses + value selects (v_cndmask), no saveexec chains.
//   - ping-pong LDS: one barrier per K-step; next tile staged into the other
//     buffer while WMMAs consume the current one.
//
// WMMA layouts per CDNA5 ISA:
//   A (16x32 f16): lane L: m = L%16, half = L/16.
//     VGPR j (pair a[2j],a[2j+1]): K = 16*(j>>2) + 8*half + 2*(j&3) + {0,1}
//   B (32x16 f16): lane L: n = L%16, half = L/16.
//     VGPR j: same K as A: 16*(j>>2) + 8*half + 2*(j&3) + {0,1}
//   C/D (16x16 f32): lane L: n = L%16; VGPR i: m = i + 8*half
//
// Input pointer order assumption (compile-only, not verifiable here):
// top-level setup_inputs() order, nested dicts flattened with sorted keys
// (JAX pytree convention) -> 54 leaves, indices documented below.
// ---------------------------------------------------------------------------

typedef __attribute__((ext_vector_type(16))) _Float16 v16h;
typedef __attribute__((ext_vector_type(8)))  float    v8f;
typedef __attribute__((ext_vector_type(2)))  __fp16   h2v;  // matches cvt_pkrtz

#define D_MODEL 768
#define NHEAD   12
#define DHEAD   64
#define FFDIM   3072
#define BATCH   4
#define SRCLEN  256
#define MAXLEN_ 64
#define TDEC_   62
#define NLAYER  6
#define NEGBIG  (-1.0e9f)

#define LSTR 18   // LDS row stride in u32 (16 pairs + 2 pad, keeps u64 align)

// -------------------------------- helpers ----------------------------------

__device__ __forceinline__ unsigned pack2h(float a, float b) {
  // single v_cvt_pkrtz_f16_f32: packs (a,b) -> {lo,hi} f16 in one register
  union { h2v h; unsigned u; } c;
  c.h = __builtin_amdgcn_cvt_pkrtz(a, b);
  return c.u;
}

__device__ __forceinline__ float gelu_tanh(float x) {
  const float c0 = 0.7978845608028654f;   // sqrt(2/pi)
  const float c1 = 0.044715f;
  float t = tanhf(c0 * (x + c1 * x * x * x));
  return 0.5f * x * (1.0f + t);
}

__device__ __forceinline__ float block_reduce_sum256(float v, float* sbuf) {
  int tid = threadIdx.x;
  sbuf[tid] = v;
  __syncthreads();
  for (int o = 128; o > 0; o >>= 1) {
    if (tid < o) sbuf[tid] += sbuf[tid + o];
    __syncthreads();
  }
  float r = sbuf[0];
  __syncthreads();
  return r;
}

__device__ __forceinline__ int imin(int a, int b) { return a < b ? a : b; }

// --------------------- tile staging (global -> LDS f16) --------------------
// Stage a 64x32 A-tile and 32x64 B-tile (as f16 pairs along K) for K-step k0.

template <bool GUARD, bool TRANSB>
__device__ __forceinline__ void
stage_tiles(unsigned* __restrict__ AshB, unsigned* __restrict__ BshB,
            const float* __restrict__ Ab, const float* __restrict__ Bb,
            int tileM, int tileN, int M, int N, int K,
            int lda, int ldb, int k0, int tid) {
  int rm  = tid >> 3;        // 0..31 : row within half-tile
  int rkq = tid & 7;         // 0..7  : float4 group (covers pairs 2q,2q+1)
  // ---------------- A tile ----------------
#pragma unroll
  for (int i = 0; i < 2; ++i) {
    int m  = rm + i * 32;
    int gm = tileM + m;
    int gk = k0 + 4 * rkq;
    uint2 p;
    if (!GUARD) {
      const float4 f = *(const float4*)(Ab + (long long)gm * lda + gk);
      p.x = pack2h(f.x, f.y);
      p.y = pack2h(f.z, f.w);
    } else {
      int cm = imin(gm, M - 1);
      const float* row = Ab + (long long)cm * lda;
      float f0 = row[imin(gk + 0, K - 1)];
      float f1 = row[imin(gk + 1, K - 1)];
      float f2 = row[imin(gk + 2, K - 1)];
      float f3 = row[imin(gk + 3, K - 1)];
      bool mo = gm >= M;
      if (mo || gk + 0 >= K) f0 = 0.f;
      if (mo || gk + 1 >= K) f1 = 0.f;
      if (mo || gk + 2 >= K) f2 = 0.f;
      if (mo || gk + 3 >= K) f3 = 0.f;
      p.x = pack2h(f0, f1);
      p.y = pack2h(f2, f3);
    }
    *(uint2*)&AshB[m * LSTR + 2 * rkq] = p;   // ds_store_b64
  }
  // ---------------- B tile ----------------
  if (TRANSB) {
    // B[N,K]: pairs contiguous along K, same mapping as A.
#pragma unroll
    for (int i = 0; i < 2; ++i) {
      int n  = rm + i * 32;
      int gn = tileN + n;
      int gk = k0 + 4 * rkq;
      uint2 p;
      if (!GUARD) {
        const float4 f = *(const float4*)(Bb + (long long)gn * ldb + gk);
        p.x = pack2h(f.x, f.y);
        p.y = pack2h(f.z, f.w);
      } else {
        int cn = imin(gn, N - 1);
        const float* row = Bb + (long long)cn * ldb;
        float f0 = row[imin(gk + 0, K - 1)];
        float f1 = row[imin(gk + 1, K - 1)];
        float f2 = row[imin(gk + 2, K - 1)];
        float f3 = row[imin(gk + 3, K - 1)];
        bool no = gn >= N;
        if (no || gk + 0 >= K) f0 = 0.f;
        if (no || gk + 1 >= K) f1 = 0.f;
        if (no || gk + 2 >= K) f2 = 0.f;
        if (no || gk + 3 >= K) f3 = 0.f;
        p.x = pack2h(f0, f1);
        p.y = pack2h(f2, f3);
      }
      *(uint2*)&BshB[n * LSTR + 2 * rkq] = p;
    }
  } else {
    // B[K,N]: load two K-rows as float4 along n, transpose-pack pairs.
    int bkp = tid >> 4;           // k-pair 0..15
    int bn0 = (tid & 15) * 4;     // 4 consecutive n
    int gk  = k0 + 2 * bkp;
    if (!GUARD) {
      const float4 r0 = *(const float4*)(Bb + (long long)gk * ldb + tileN + bn0);
      const float4 r1 = *(const float4*)(Bb + (long long)(gk + 1) * ldb + tileN + bn0);
      BshB[(bn0 + 0) * LSTR + bkp] = pack2h(r0.x, r1.x);
      BshB[(bn0 + 1) * LSTR + bkp] = pack2h(r0.y, r1.y);
      BshB[(bn0 + 2) * LSTR + bkp] = pack2h(r0.z, r1.z);
      BshB[(bn0 + 3) * LSTR + bkp] = pack2h(r0.w, r1.w);
    } else {
      int ck0 = imin(gk, K - 1);
      int ck1 = imin(gk + 1, K - 1);
      bool k0ok = gk < K;
      bool k1ok = gk + 1 < K;
#pragma unroll
      for (int j = 0; j < 4; ++j) {
        int gn = tileN + bn0 + j;
        int cn = imin(gn, N - 1);
        float f0 = Bb[(long long)ck0 * ldb + cn];
        float f1 = Bb[(long long)ck1 * ldb + cn];
        if (gn >= N || !k0ok) f0 = 0.f;
        if (gn >= N || !k1ok) f1 = 0.f;
        BshB[(bn0 + j) * LSTR + bkp] = pack2h(f0, f1);
      }
    }
  }
}

// ------------------------- generic WMMA GEMM kernel ------------------------
// C[z] = act(alpha * A[z] @ op(B)[z] + bias), batched via blockIdx.z with
// two-level offsets: off = (z/zdiv)*o?1 + (z%zdiv)*o?2.
// A: row-major [M,K] stride lda. B: row-major [K,N] stride ldb, or if
// TRANSB: row-major [N,K] stride ldb.
// GUARD=false requires M%64==0, N%64==0, K%32==0 (checked by the host).

template <bool GUARD, bool TRANSB>
__global__ void __launch_bounds__(256)
gemm_wmma_kernel(const float* __restrict__ A, const float* __restrict__ Bm,
                 const float* __restrict__ bias, float* __restrict__ C,
                 int M, int N, int K, int lda, int ldb, int ldc,
                 long long oA1, long long oA2, long long oB1, long long oB2,
                 long long oC1, long long oC2,
                 int zdiv, int act, float alpha) {
  __shared__ unsigned Ash[2][64 * LSTR];   // ping-pong A tiles
  __shared__ unsigned Bsh[2][64 * LSTR];   // ping-pong B tiles

  int z  = blockIdx.z;
  int zb = z / zdiv;
  int zh = z % zdiv;
  const float* Ab = A  + zb * oA1 + zh * oA2;
  const float* Bb = Bm + zb * oB1 + zh * oB2;
  float*       Cb = C  + zb * oC1 + zh * oC2;

  int tileM = blockIdx.x * 64;
  int tileN = blockIdx.y * 64;

  int tid  = threadIdx.x;
  int lane = tid & 31;
  int wid  = tid >> 5;     // 8 waves
  int wrow = wid >> 2;     // 0..1 : 32-row strip
  int wcol = wid & 3;      // 0..3 : 16-col strip
  int half = lane >> 4;    // lane half (wave32)
  int l16  = lane & 15;

  v8f acc0 = {};
  v8f acc1 = {};

  int nsteps = (K + 31) >> 5;
  stage_tiles<GUARD, TRANSB>(Ash[0], Bsh[0], Ab, Bb, tileM, tileN,
                             M, N, K, lda, ldb, 0, tid);

  int ar0 = (wrow * 32 + l16) * LSTR;
  int ar1 = ar0 + 16 * LSTR;
  int br  = (wcol * 16 + l16) * LSTR;

  for (int s = 0; s < nsteps; ++s) {
    int cur = s & 1;
    __syncthreads();
    if (s + 1 < nsteps)
      stage_tiles<GUARD, TRANSB>(Ash[cur ^ 1], Bsh[cur ^ 1], Ab, Bb,
                                 tileM, tileN, M, N, K, lda, ldb,
                                 (s + 1) * 32, tid);
    if (!GUARD && s + 2 < nsteps)   // prefetch A for step s+2
      __builtin_prefetch(Ab + (long long)(tileM + (tid & 63)) * lda +
                             (s + 2) * 32, 0, 1);

    // ---- build fragments and issue WMMA ----
    union { v16h h; unsigned u[8]; } fa0, fa1, fb;
    const unsigned* As = Ash[cur];
    const unsigned* Bs = Bsh[cur];
#pragma unroll
    for (int j = 0; j < 8; ++j) {
      int apair = ((j >> 2) << 3) + (half << 2) + (j & 3);  // A layout
      fa0.u[j] = As[ar0 + apair];
      fa1.u[j] = As[ar1 + apair];
      fb.u[j]  = Bs[br + apair];                            // B layout
    }
    acc0 = __builtin_amdgcn_wmma_f32_16x16x32_f16(
        false, fa0.h, false, fb.h, (short)0, acc0, false, false);
    acc1 = __builtin_amdgcn_wmma_f32_16x16x32_f16(
        false, fa1.h, false, fb.h, (short)0, acc1, false, false);
  }

  // ---- epilogue: alpha, bias, activation, (guarded) store ----
  int gn = tileN + wcol * 16 + l16;
  if (GUARD && gn >= N) return;
  float bval = bias ? bias[gn] : 0.f;
  union { v8f v; float f[8]; } r0, r1;
  r0.v = acc0;
  r1.v = acc1;
#pragma unroll
  for (int i = 0; i < 8; ++i) {
    int gm0 = tileM + wrow * 32 + half * 8 + i;   // C/D layout: m = i + 8*half
    if (!GUARD || gm0 < M) {
      float v = alpha * r0.f[i] + bval;
      if (act == 1) v = gelu_tanh(v);
      Cb[(long long)gm0 * ldc + gn] = v;
    }
    int gm1 = gm0 + 16;
    if (!GUARD || gm1 < M) {
      float v = alpha * r1.f[i] + bval;
      if (act == 1) v = gelu_tanh(v);
      Cb[(long long)gm1 * ldc + gn] = v;
    }
  }
}

// ---------------------- embedding + LayerNorm kernel -----------------------
// row = b*T + t ; x = LN(tok_emb[ids[b,t]] + pos_emb[t]) ; D=768, 256 thr.

__global__ void __launch_bounds__(256)
embed_ln_kernel(const int* __restrict__ ids, int ids_stride, int T,
                const float* __restrict__ tok_emb, const float* __restrict__ pos_emb,
                const float* __restrict__ gamma, const float* __restrict__ beta,
                float* __restrict__ out) {
  __shared__ float sbuf[256];
  int row = blockIdx.x;
  int b = row / T, t = row % T;
  int id = ids[b * ids_stride + t];
  const float* te = tok_emb + (long long)id * D_MODEL;
  const float* pe = pos_emb + (long long)t * D_MODEL;
  int tid = threadIdx.x;
  float loc[3];
  float s = 0.f;
#pragma unroll
  for (int i = 0; i < 3; ++i) {
    int c = tid + i * 256;
    float v = te[c] + pe[c];
    loc[i] = v;
    s += v;
  }
  float mean = block_reduce_sum256(s, sbuf) * (1.0f / D_MODEL);
  float vs = 0.f;
#pragma unroll
  for (int i = 0; i < 3; ++i) { float d = loc[i] - mean; vs += d * d; }
  float var  = block_reduce_sum256(vs, sbuf) * (1.0f / D_MODEL);
  float rstd = rsqrtf(var + 1e-5f);
  float* orow = out + (long long)row * D_MODEL;
#pragma unroll
  for (int i = 0; i < 3; ++i) {
    int c = tid + i * 256;
    orow[c] = gamma[c] * (loc[i] - mean) * rstd + beta[c];
  }
}

// ------------------------- residual + LayerNorm ----------------------------
// out = LN(res + proj) ; out may alias res (each element touched by 1 thread)

__global__ void __launch_bounds__(256)
residual_ln_kernel(const float* res, const float* proj,
                   const float* __restrict__ gamma, const float* __restrict__ beta,
                   float* out) {
  __shared__ float sbuf[256];
  int row = blockIdx.x;
  int tid = threadIdx.x;
  const float* rr = res  + (long long)row * D_MODEL;
  const float* pp = proj + (long long)row * D_MODEL;
  float loc[3];
  float s = 0.f;
#pragma unroll
  for (int i = 0; i < 3; ++i) {
    int c = tid + i * 256;
    float v = rr[c] + pp[c];
    loc[i] = v;
    s += v;
  }
  float mean = block_reduce_sum256(s, sbuf) * (1.0f / D_MODEL);
  float vs = 0.f;
#pragma unroll
  for (int i = 0; i < 3; ++i) { float d = loc[i] - mean; vs += d * d; }
  float var  = block_reduce_sum256(vs, sbuf) * (1.0f / D_MODEL);
  float rstd = rsqrtf(var + 1e-5f);
  float* oo = out + (long long)row * D_MODEL;
#pragma unroll
  for (int i = 0; i < 3; ++i) {
    int c = tid + i * 256;
    oo[c] = gamma[c] * (loc[i] - mean) * rstd + beta[c];
  }
}

// ---------------- softmax with scale + pad-mask + causal bias --------------
// S[z, q, :Tk] : blockIdx.x=q, blockIdx.y=z, b = z/hdiv. Tk <= 256.

__global__ void __launch_bounds__(256)
softmax_rows_kernel(float* __restrict__ S, const int* __restrict__ mask,
                    int mask_stride, int Tq, int Tk, int hdiv,
                    float scale, int causal) {
  __shared__ float sbuf[256];
  int q = blockIdx.x;
  int z = blockIdx.y;
  int b = z / hdiv;
  float* row = S + ((long long)z * Tq + q) * Tk;
  int tid = threadIdx.x;
  bool active = tid < Tk;
  float val = -3.0e38f;
  if (active) {
    val = row[tid] * scale;
    if (mask && mask[b * mask_stride + tid] == 0) val += NEGBIG;
    if (causal && tid > q) val += NEGBIG;
  }
  sbuf[tid] = val;
  __syncthreads();
  for (int o = 128; o > 0; o >>= 1) {
    if (tid < o) sbuf[tid] = fmaxf(sbuf[tid], sbuf[tid + o]);
    __syncthreads();
  }
  float mx = sbuf[0];
  __syncthreads();
  float e = active ? __expf(val - mx) : 0.f;
  float ssum = block_reduce_sum256(e, sbuf);
  float inv = 1.0f / ssum;
  if (active) row[tid] = e * inv;
}

// ------------------------------ host driver --------------------------------

extern "C" void kernel_launch(void* const* d_in, const int* in_sizes, int n_in,
                              void* d_out, int out_size, void* d_ws, size_t ws_size,
                              hipStream_t stream) {
  (void)in_sizes; (void)n_in; (void)out_size; (void)ws_size;

  // --- input leaves (top-level order; nested dicts sorted-key flattened) ---
  const int*   enc_ids  = (const int*)d_in[0];
  const int*   enc_mask = (const int*)d_in[1];
  const int*   dec_ids  = (const int*)d_in[2];
  const int*   dec_mask = (const int*)d_in[3];
  const float* tok_emb  = (const float*)d_in[4];
  const float* pos_enc  = (const float*)d_in[5];
  const float* pos_dec  = (const float*)d_in[6];
  // emb_ln sorted: db, dg, eb, eg
  const float* emb_db = (const float*)d_in[7];
  const float* emb_dg = (const float*)d_in[8];
  const float* emb_eb = (const float*)d_in[9];
  const float* emb_eg = (const float*)d_in[10];
  // enc_layers sorted: W1 W2 Wk Wo Wq Wv b1 b2 bk bo bq bv ln1_b ln1_g ln2_b ln2_g
  const float* eW1  = (const float*)d_in[11];
  const float* eW2  = (const float*)d_in[12];
  const float* eWk  = (const float*)d_in[13];
  const float* eWo  = (const float*)d_in[14];
  const float* eWq  = (const float*)d_in[15];
  const float* eWv  = (const float*)d_in[16];
  const float* eb1  = (const float*)d_in[17];
  const float* eb2  = (const float*)d_in[18];
  const float* ebk  = (const float*)d_in[19];
  const float* ebo  = (const float*)d_in[20];
  const float* ebq  = (const float*)d_in[21];
  const float* ebv  = (const float*)d_in[22];
  const float* eln1b = (const float*)d_in[23];
  const float* eln1g = (const float*)d_in[24];
  const float* eln2b = (const float*)d_in[25];
  const float* eln2g = (const float*)d_in[26];
  // dec_layers sorted: W1 W2 Wk Wo Wq Wv b1 b2 bk bo bq bv cWk cWo cWq cWv
  //                    cbk cbo cbq cbv ln1_b ln1_g ln2_b ln2_g ln3_b ln3_g
  const float* dW1  = (const float*)d_in[27];
  const float* dW2  = (const float*)d_in[28];
  const float* dWk  = (const float*)d_in[29];
  const float* dWo  = (const float*)d_in[30];
  const float* dWq  = (const float*)d_in[31];
  const float* dWv  = (const float*)d_in[32];
  const float* db1  = (const float*)d_in[33];
  const float* db2  = (const float*)d_in[34];
  const float* dbk  = (const float*)d_in[35];
  const float* dbo  = (const float*)d_in[36];
  const float* dbq  = (const float*)d_in[37];
  const float* dbv  = (const float*)d_in[38];
  const float* dcWk = (const float*)d_in[39];
  const float* dcWo = (const float*)d_in[40];
  const float* dcWq = (const float*)d_in[41];
  const float* dcWv = (const float*)d_in[42];
  const float* dcbk = (const float*)d_in[43];
  const float* dcbo = (const float*)d_in[44];
  const float* dcbq = (const float*)d_in[45];
  const float* dcbv = (const float*)d_in[46];
  const float* dln1b = (const float*)d_in[47];
  const float* dln1g = (const float*)d_in[48];
  const float* dln2b = (const float*)d_in[49];
  const float* dln2g = (const float*)d_in[50];
  const float* dln3b = (const float*)d_in[51];
  const float* dln3g = (const float*)d_in[52];
  const float* copyW = (const float*)d_in[53];

  float* out = (float*)d_out;

  // --------------------------- workspace carve-up --------------------------
  float* ws = (float*)d_ws;
  size_t off = 0;
  float* Xe  = ws + off; off += (size_t)BATCH * SRCLEN * D_MODEL;    // enc hidden
  float* Xd  = ws + off; off += (size_t)BATCH * MAXLEN_ * D_MODEL;   // dec hidden (248 rows used)
  float* Qb  = ws + off; off += (size_t)BATCH * SRCLEN * D_MODEL;
  float* Kb  = ws + off; off += (size_t)BATCH * SRCLEN * D_MODEL;
  float* Vb  = ws + off; off += (size_t)BATCH * SRCLEN * D_MODEL;
  float* Sb  = ws + off; off += (size_t)BATCH * NHEAD * SRCLEN * SRCLEN;
  float* Ob  = ws + off; off += (size_t)BATCH * SRCLEN * D_MODEL;
  float* Pb  = ws + off; off += (size_t)BATCH * SRCLEN * D_MODEL;
  float* Mid = ws + off; off += (size_t)BATCH * SRCLEN * FFDIM;
  float* T1  = ws + off; off += (size_t)BATCH * TDEC_ * D_MODEL;

  const float SCALE      = 0.125f;                 // 1/sqrt(64)
  const float COPY_SCALE = 0.03608439182435161f;   // 1/sqrt(768)

  auto gemm = [&](const float* A, const float* Bm, const float* bias, float* C,
                  int M, int N, int K, int lda, int ldb, int ldc,
                  long long oA1, long long oA2, long long oB1, long long oB2,
                  long long oC1, long long oC2, int zdiv, int nbatch,
                  int transB, int act, float alpha) {
    dim3 grid((M + 63) / 64, (N + 63) / 64, nbatch);
    dim3 blk(256);
    bool aligned = (M % 64 == 0) && (N % 64 == 0) && (K % 32 == 0);
    if (transB) {
      if (aligned)
        gemm_wmma_kernel<false, true><<<grid, blk, 0, stream>>>(
            A, Bm, bias, C, M, N, K, lda, ldb, ldc,
            oA1, oA2, oB1, oB2, oC1, oC2, zdiv, act, alpha);
      else
        gemm_wmma_kernel<true, true><<<grid, blk, 0, stream>>>(
            A, Bm, bias, C, M, N, K, lda, ldb, ldc,
            oA1, oA2, oB1, oB2, oC1, oC2, zdiv, act, alpha);
    } else {
      if (aligned)
        gemm_wmma_kernel<false, false><<<grid, blk, 0, stream>>>(
            A, Bm, bias, C, M, N, K, lda, ldb, ldc,
            oA1, oA2, oB1, oB2, oC1, oC2, zdiv, act, alpha);
      else
        gemm_wmma_kernel<true, false><<<grid, blk, 0, stream>>>(
            A, Bm, bias, C, M, N, K, lda, ldb, ldc,
            oA1, oA2, oB1, oB2, oC1, oC2, zdiv, act, alpha);
    }
  };
  auto lnres = [&](const float* res, const float* proj, const float* g,
                   const float* b, float* o2, int rows) {
    residual_ln_kernel<<<dim3(rows), dim3(256), 0, stream>>>(res, proj, g, b, o2);
  };
  auto softmax = [&](float* S, const int* mask, int mstride, int Tq, int Tk,
                     int hdiv, float scale, int causal, int zcnt) {
    softmax_rows_kernel<<<dim3(Tq, zcnt), dim3(256), 0, stream>>>(
        S, mask, mstride, Tq, Tk, hdiv, scale, causal);
  };

  const int Me = BATCH * SRCLEN;   // 1024 encoder rows
  const int Md = BATCH * TDEC_;    // 248 decoder rows
  const long long sdQe = (long long)SRCLEN * D_MODEL;   // per-batch stride enc
  const long long sdQd = (long long)TDEC_  * D_MODEL;   // per-batch stride dec

  // ================================ ENCODER ================================
  embed_ln_kernel<<<dim3(Me), dim3(256), 0, stream>>>(
      enc_ids, SRCLEN, SRCLEN, tok_emb, pos_enc, emb_eg, emb_eb, Xe);

  for (int l = 0; l < NLAYER; ++l) {
    size_t oDD = (size_t)l * D_MODEL * D_MODEL;
    size_t oD  = (size_t)l * D_MODEL;
    size_t oDF = (size_t)l * D_MODEL * FFDIM;
    size_t oF  = (size_t)l * FFDIM;

    gemm(Xe, eWq + oDD, ebq + oD, Qb, Me, D_MODEL, D_MODEL,
         D_MODEL, D_MODEL, D_MODEL, 0,0,0,0,0,0, 1, 1, 0, 0, 1.f);
    gemm(Xe, eWk + oDD, ebk + oD, Kb, Me, D_MODEL, D_MODEL,
         D_MODEL, D_MODEL, D_MODEL, 0,0,0,0,0,0, 1, 1, 0, 0, 1.f);
    gemm(Xe, eWv + oDD, ebv + oD, Vb, Me, D_MODEL, D_MODEL,
         D_MODEL, D_MODEL, D_MODEL, 0,0,0,0,0,0, 1, 1, 0, 0, 1.f);
    // S = Q @ K^T  (batched over b*H)
    gemm(Qb, Kb, nullptr, Sb, SRCLEN, SRCLEN, DHEAD,
         D_MODEL, D_MODEL, SRCLEN,
         sdQe, DHEAD, sdQe, DHEAD,
         (long long)NHEAD * SRCLEN * SRCLEN, (long long)SRCLEN * SRCLEN,
         NHEAD, BATCH * NHEAD, 1, 0, 1.f);
    softmax(Sb, enc_mask, SRCLEN, SRCLEN, SRCLEN, NHEAD, SCALE, 0, BATCH * NHEAD);
    // O = A @ V
    gemm(Sb, Vb, nullptr, Ob, SRCLEN, DHEAD, SRCLEN,
         SRCLEN, D_MODEL, D_MODEL,
         (long long)NHEAD * SRCLEN * SRCLEN, (long long)SRCLEN * SRCLEN,
         sdQe, DHEAD, sdQe, DHEAD,
         NHEAD, BATCH * NHEAD, 0, 0, 1.f);
    gemm(Ob, eWo + oDD, ebo + oD, Pb, Me, D_MODEL, D_MODEL,
         D_MODEL, D_MODEL, D_MODEL, 0,0,0,0,0,0, 1, 1, 0, 0, 1.f);
    lnres(Xe, Pb, eln1g + oD, eln1b + oD, Xe, Me);
    // FFN
    gemm(Xe, eW1 + oDF, eb1 + oF, Mid, Me, FFDIM, D_MODEL,
         D_MODEL, FFDIM, FFDIM, 0,0,0,0,0,0, 1, 1, 0, 1, 1.f);
    gemm(Mid, eW2 + oDF, eb2 + oD, Pb, Me, D_MODEL, FFDIM,
         FFDIM, D_MODEL, D_MODEL, 0,0,0,0,0,0, 1, 1, 0, 0, 1.f);
    lnres(Xe, Pb, eln2g + oD, eln2b + oD, Xe, Me);
  }

  // ================================ DECODER ================================
  embed_ln_kernel<<<dim3(Md), dim3(256), 0, stream>>>(
      dec_ids, MAXLEN_, TDEC_, tok_emb, pos_dec, emb_dg, emb_db, Xd);

  for (int l = 0; l < NLAYER; ++l) {
    size_t oDD = (size_t)l * D_MODEL * D_MODEL;
    size_t oD  = (size_t)l * D_MODEL;
    size_t oDF = (size_t)l * D_MODEL * FFDIM;
    size_t oF  = (size_t)l * FFDIM;

    // ---- masked self-attention ----
    gemm(Xd, dWq + oDD, dbq + oD, Qb, Md, D_MODEL, D_MODEL,
         D_MODEL, D_MODEL, D_MODEL, 0,0,0,0,0,0, 1, 1, 0, 0, 1.f);
    gemm(Xd, dWk + oDD, dbk + oD, Kb, Md, D_MODEL, D_MODEL,
         D_MODEL, D_MODEL, D_MODEL, 0,0,0,0,0,0, 1, 1, 0, 0, 1.f);
    gemm(Xd, dWv + oDD, dbv + oD, Vb, Md, D_MODEL, D_MODEL,
         D_MODEL, D_MODEL, D_MODEL, 0,0,0,0,0,0, 1, 1, 0, 0, 1.f);
    gemm(Qb, Kb, nullptr, Sb, TDEC_, TDEC_, DHEAD,
         D_MODEL, D_MODEL, TDEC_,
         sdQd, DHEAD, sdQd, DHEAD,
         (long long)NHEAD * TDEC_ * TDEC_, (long long)TDEC_ * TDEC_,
         NHEAD, BATCH * NHEAD, 1, 0, 1.f);
    softmax(Sb, dec_mask, MAXLEN_, TDEC_, TDEC_, NHEAD, SCALE, 1, BATCH * NHEAD);
    gemm(Sb, Vb, nullptr, Ob, TDEC_, DHEAD, TDEC_,
         TDEC_, D_MODEL, D_MODEL,
         (long long)NHEAD * TDEC_ * TDEC_, (long long)TDEC_ * TDEC_,
         sdQd, DHEAD, sdQd, DHEAD,
         NHEAD, BATCH * NHEAD, 0, 0, 1.f);
    gemm(Ob, dWo + oDD, dbo + oD, Pb, Md, D_MODEL, D_MODEL,
         D_MODEL, D_MODEL, D_MODEL, 0,0,0,0,0,0, 1, 1, 0, 0, 1.f);
    lnres(Xd, Pb, dln1g + oD, dln1b + oD, Xd, Md);

    // ---- cross-attention ----
    gemm(Xd, dcWq + oDD, dcbq + oD, Qb, Md, D_MODEL, D_MODEL,
         D_MODEL, D_MODEL, D_MODEL, 0,0,0,0,0,0, 1, 1, 0, 0, 1.f);
    gemm(Xe, dcWk + oDD, dcbk + oD, Kb, Me, D_MODEL, D_MODEL,
         D_MODEL, D_MODEL, D_MODEL, 0,0,0,0,0,0, 1, 1, 0, 0, 1.f);
    gemm(Xe, dcWv + oDD, dcbv + oD, Vb, Me, D_MODEL, D_MODEL,
         D_MODEL, D_MODEL, D_MODEL, 0,0,0,0,0,0, 1, 1, 0, 0, 1.f);
    gemm(Qb, Kb, nullptr, Sb, TDEC_, SRCLEN, DHEAD,
         D_MODEL, D_MODEL, SRCLEN,
         sdQd, DHEAD, sdQe, DHEAD,
         (long long)NHEAD * TDEC_ * SRCLEN, (long long)TDEC_ * SRCLEN,
         NHEAD, BATCH * NHEAD, 1, 0, 1.f);
    softmax(Sb, enc_mask, SRCLEN, TDEC_, SRCLEN, NHEAD, SCALE, 0, BATCH * NHEAD);
    gemm(Sb, Vb, nullptr, Ob, TDEC_, DHEAD, SRCLEN,
         SRCLEN, D_MODEL, D_MODEL,
         (long long)NHEAD * TDEC_ * SRCLEN, (long long)TDEC_ * SRCLEN,
         sdQe, DHEAD, sdQd, DHEAD,
         NHEAD, BATCH * NHEAD, 0, 0, 1.f);
    gemm(Ob, dcWo + oDD, dcbo + oD, Pb, Md, D_MODEL, D_MODEL,
         D_MODEL, D_MODEL, D_MODEL, 0,0,0,0,0,0, 1, 1, 0, 0, 1.f);
    lnres(Xd, Pb, dln3g + oD, dln3b + oD, Xd, Md);

    // ---- FFN ----
    gemm(Xd, dW1 + oDF, db1 + oF, Mid, Md, FFDIM, D_MODEL,
         D_MODEL, FFDIM, FFDIM, 0,0,0,0,0,0, 1, 1, 0, 1, 1.f);
    gemm(Mid, dW2 + oDF, db2 + oD, Pb, Md, D_MODEL, FFDIM,
         FFDIM, D_MODEL, D_MODEL, 0,0,0,0,0,0, 1, 1, 0, 0, 1.f);
    lnres(Xd, Pb, dln2g + oD, dln2b + oD, Xd, Md);
  }

  // ============================ copy-pointer head ==========================
  // T1 = dec @ copy_W
  gemm(Xd, copyW, nullptr, T1, Md, D_MODEL, D_MODEL,
       D_MODEL, D_MODEL, D_MODEL, 0,0,0,0,0,0, 1, 1, 0, 0, 1.f);
  // scores[b] = T1[b] @ enc[b]^T  (per-batch), raw into d_out
  gemm(T1, Xe, nullptr, out, TDEC_, SRCLEN, D_MODEL,
       D_MODEL, D_MODEL, SRCLEN,
       sdQd, 0, sdQe, 0, (long long)TDEC_ * SRCLEN, 0,
       1, BATCH, 1, 0, 1.f);
  // softmax(scores * COPY_SCALE + src_pad * NEG)
  softmax(out, enc_mask, SRCLEN, TDEC_, SRCLEN, 1, COPY_SCALE, 0, BATCH);
}
